// Volume_construct_53996328845953
// MI455X (gfx1250) — compile-verified, matrix operand order checked by reading, not run on hardware
//
#include <hip/hip_runtime.h>

// ---------------------------------------------------------------------------
// Problem constants
// ---------------------------------------------------------------------------
#define BATCH 8
#define CIN0  320
#define HGT   128
#define WID   256
#define NDISP 48
#define HWPIX (HGT*WID)                    // 32768
#define NPIX  ((size_t)BATCH*HWPIX)        // 262144
#define EPSBN 1e-5f

typedef __attribute__((ext_vector_type(16))) _Float16 v16h;
typedef __attribute__((ext_vector_type(8)))  _Float16 v8h;
typedef __attribute__((ext_vector_type(8)))  float    v8f;

union BU { v16h v; unsigned u[8]; };
union HU { unsigned u; _Float16 h[2]; };

// K index base for A/B fragment dword v (holds halves K=kb, kb+1) per the
// CDNA5 16-bit A-matrix layout: VGPR0..3 -> K{0..7}(+8 for hi lanes),
// VGPR4..7 -> K{16..23}(+8 for hi lanes).
__device__ __forceinline__ int kbase(int v, int hi) {
  int b = (v < 4) ? (2 * v) : (16 + 2 * (v - 4));
  return b + (hi ? 8 : 0);
}

// ---------------------------------------------------------------------------
// Workspace layout (bytes)
// ---------------------------------------------------------------------------
constexpr size_t OFF_SWZ0  = 0;                          // 256x320 f16 swizzled
constexpr size_t OFF_SWZ1  = OFF_SWZ0  + 81920ull * 2;   // 128x256
constexpr size_t OFF_SWZ2  = OFF_SWZ1  + 32768ull * 2;   // 64x128
constexpr size_t OFF_SWZ3  = OFF_SWZ2  + 8192ull  * 2;   // 32x64
constexpr size_t OFF_SWZ4E = OFF_SWZ3  + 2048ull  * 2;   // 32x32 (even in-ch)
constexpr size_t OFF_SWZ4O = OFF_SWZ4E + 1024ull  * 2;   // 32x32 (odd in-ch)
constexpr size_t OFF_SWZ5  = OFF_SWZ4O + 1024ull  * 2;   // 16x32
constexpr size_t OFF_BIAS  = OFF_SWZ5  + 512ull   * 2;   // = 254976
constexpr size_t OFF_B0 = OFF_BIAS + 0;      // 256 f32
constexpr size_t OFF_B1 = OFF_BIAS + 1024;   // 128
constexpr size_t OFF_B2 = OFF_BIAS + 1536;   // 64
constexpr size_t OFF_B3 = OFF_BIAS + 1792;   // 32
constexpr size_t OFF_B4 = OFF_BIAS + 1920;   // 32
constexpr size_t OFF_BZ = OFF_BIAS + 2048;   // 32 (zeros for Q)
constexpr size_t OFF_B5 = OFF_BIAS + 2176;   // 16
constexpr size_t OFF_W6 = OFF_BIAS + 2240;   // 16 f32 (folded 16->1 weights)
constexpr size_t OFF_B6 = OFF_BIAS + 2304;   // 1
constexpr size_t OFF_BUF = OFF_BIAS + 4096;
constexpr size_t OFF_Y1  = OFF_BUF;                      // [NPIX][256] f16
constexpr size_t OFF_Y2  = OFF_Y1  + NPIX * 256 * 2;     // [NPIX][128] f16
constexpr size_t OFF_F64 = OFF_Y2  + NPIX * 128 * 2;     // [NPIX][64]  f16
constexpr size_t OFF_F2L = OFF_F64 + NPIX * 64  * 2;     // [NPIX][32]  f16
constexpr size_t OFF_F2R = OFF_F2L + NPIX * 32  * 2;
constexpr size_t OFF_P   = OFF_F2R + NPIX * 32  * 2;
constexpr size_t OFF_Q   = OFF_P   + NPIX * 32  * 2;

// ---------------------------------------------------------------------------
// Weight prep: fold BN scale into W, convert to f16, swizzle into the exact
// per-lane WMMA A-fragment layout so a fragment is one 32B load per lane.
// cstep/coff select even/odd input channels for the cost-volume split.
// ---------------------------------------------------------------------------
__global__ void prep_wb(const float* __restrict__ w, const float* __restrict__ gma,
                        const float* __restrict__ bta, const float* __restrict__ mn,
                        const float* __restrict__ vr,
                        int COUT, int CINW, int CIN_SRC, int cstep, int coff,
                        _Float16* __restrict__ dstw, float* __restrict__ dstb,
                        int write_bias) {
  const int KT = CINW >> 5;
  const int total = COUT * CINW;
  const int nth = gridDim.x * blockDim.x;
  for (int s = blockIdx.x * blockDim.x + threadIdx.x; s < total; s += nth) {
    int lane_h = s & 15;
    int lane   = (s >> 4) & 31;
    int kt     = (s >> 9) % KT;
    int mt     = s / (512 * KT);
    int v  = lane_h >> 1;
    int k  = kt * 32 + kbase(v, lane & 16) + (lane_h & 1);
    int m  = mt * 16 + (lane & 15);
    float inv = gma[m] * rsqrtf(vr[m] + EPSBN);
    dstw[s] = (_Float16)(w[(size_t)m * CIN_SRC + k * cstep + coff] * inv);
  }
  if (write_bias) {
    for (int c = blockIdx.x * blockDim.x + threadIdx.x; c < COUT; c += nth) {
      float inv = gma[c] * rsqrtf(vr[c] + EPSBN);
      dstb[c] = bta[c] - mn[c] * inv;
    }
  }
}

__global__ void prep_small(const float* __restrict__ w6, const float* __restrict__ g,
                           const float* __restrict__ be, const float* __restrict__ mnn,
                           const float* __restrict__ vrr,
                           float* __restrict__ w6dst, float* __restrict__ b6dst,
                           float* __restrict__ zb) {
  int t = threadIdx.x;
  float inv = g[0] * rsqrtf(vrr[0] + EPSBN);
  if (t < 16) w6dst[t] = w6[t] * inv;
  if (t == 0) b6dst[0] = be[0] - mnn[0] * inv;
  if (t < 32) zb[t] = 0.f;
}

// ---------------------------------------------------------------------------
// Layer 0: fp32 channel-major input -> LDS transpose/convert -> WMMA.
// 320 -> 256, ReLU, output pixel-major f16. 64-pixel tile per block,
// 2 waves * 2 subtiles: each A fragment load feeds 2 independent WMMAs
// (halves A-fragment L0 traffic; breaks the accumulator dependency chain).
// CP = 322 (CP ≡ 2 mod 4): dword-aligned fragment reads, odd dword stride
// (161) => conflict-free LDS banks.
// ---------------------------------------------------------------------------
__global__ __launch_bounds__(64) void conv_l0(
    const float* __restrict__ X, const _Float16* __restrict__ Wswz,
    const float* __restrict__ bias, _Float16* __restrict__ Y) {
  constexpr int CIN = CIN0, COUT = 256, KT = 10, MT = 16, CP = 322, NSUB = 2;
  __shared__ _Float16 lds[64 * CP];
  const int tid = threadIdx.x;
  const size_t g0 = (size_t)blockIdx.x * 64;          // 64 | HW: tile in one b
  const int b = (int)(g0 / HWPIX);
  const size_t hw0 = g0 % HWPIX;
  const float* xb = X + (size_t)b * CIN * HWPIX + hw0;
  for (int i = tid; i < CIN * 64; i += 64) {
    int c = i >> 6, p = i & 63;
    lds[p * CP + c] = (_Float16)xb[(size_t)c * HWPIX + p];
  }
  __syncthreads();
  const int lane = tid & 31, wid = tid >> 5;
  const int nl = lane & 15, hi = lane & 16, kh = hi ? 8 : 0;
  const int p0 = wid * 32 + nl;                        // subtile s pixel: p0+16s
  v16h bfr[NSUB][KT];
#pragma unroll
  for (int s = 0; s < NSUB; ++s)
#pragma unroll
    for (int kt = 0; kt < KT; ++kt) {
      BU u;
#pragma unroll
      for (int v = 0; v < 8; ++v)
        u.u[v] = *(const unsigned*)(&lds[(p0 + s * 16) * CP + kt * 32 + kbase(v, hi)]);
      bfr[s][kt] = u.v;
    }
#pragma unroll 1
  for (int mt = 0; mt < MT; ++mt) {
    v8f acc[NSUB] = {};
#pragma unroll
    for (int kt = 0; kt < KT; ++kt) {
      v16h a = *(const v16h*)(Wswz + ((size_t)(mt * KT + kt) * 32 + lane) * 16);
#pragma unroll
      for (int s = 0; s < NSUB; ++s)
        acc[s] = __builtin_amdgcn_wmma_f32_16x16x32_f16(false, a, false, bfr[s][kt],
                                                        (short)0, acc[s], false, false);
    }
    const int ch0 = mt * 16 + kh;
#pragma unroll
    for (int s = 0; s < NSUB; ++s) {
      v8h o;
#pragma unroll
      for (int r = 0; r < 8; ++r)
        o[r] = (_Float16)fmaxf(acc[s][r] + bias[ch0 + r], 0.f);
      *(v8h*)(Y + (g0 + p0 + s * 16) * COUT + ch0) = o;
    }
  }
}

// ---------------------------------------------------------------------------
// Generic pixel-major f16 GEMM layer: Y[pix][COUT] = act(W @ act(X[pix][CIN]) + b)
// NSUB 16-pixel subtiles per wave; B fragments held in registers across the MT
// loop; each pre-swizzled A fragment feeds NSUB independent WMMAs.
// ---------------------------------------------------------------------------
template <int CIN, int COUT, int NSUB, bool RELU_IN, bool RELU_OUT>
__global__ __launch_bounds__(256) void gemm_pm(
    const _Float16* __restrict__ X, const _Float16* __restrict__ Wswz,
    const float* __restrict__ bias, _Float16* __restrict__ Y) {
  constexpr int KT = CIN / 32, MT = COUT / 16;
  const int lane = threadIdx.x & 31, wid = threadIdx.x >> 5;
  const int hi = lane & 16, kh = hi ? 8 : 0;
  const size_t pix0 = (size_t)blockIdx.x * (128 * NSUB) + wid * (16 * NSUB) + (lane & 15);
  v16h bfr[NSUB][KT];
#pragma unroll
  for (int s = 0; s < NSUB; ++s)
#pragma unroll
    for (int kt = 0; kt < KT; ++kt) {
      BU u;
#pragma unroll
      for (int v = 0; v < 8; ++v) {
        unsigned d = *(const unsigned*)(X + (pix0 + s * 16) * CIN + kt * 32 + kbase(v, hi));
        if (RELU_IN) {
          HU h; h.u = d;
          h.h[0] = h.h[0] > (_Float16)0 ? h.h[0] : (_Float16)0;
          h.h[1] = h.h[1] > (_Float16)0 ? h.h[1] : (_Float16)0;
          d = h.u;
        }
        u.u[v] = d;
      }
      bfr[s][kt] = u.v;
    }
#pragma unroll 1
  for (int mt = 0; mt < MT; ++mt) {
    v8f acc[NSUB] = {};
#pragma unroll
    for (int kt = 0; kt < KT; ++kt) {
      v16h a = *(const v16h*)(Wswz + ((size_t)(mt * KT + kt) * 32 + lane) * 16);
#pragma unroll
      for (int s = 0; s < NSUB; ++s)
        acc[s] = __builtin_amdgcn_wmma_f32_16x16x32_f16(false, a, false, bfr[s][kt],
                                                        (short)0, acc[s], false, false);
    }
    const int ch0 = mt * 16 + kh;
#pragma unroll
    for (int s = 0; s < NSUB; ++s) {
      v8h o;
#pragma unroll
      for (int r = 0; r < 8; ++r) {
        float y = acc[s][r] + bias[ch0 + r];
        if (RELU_OUT) y = fmaxf(y, 0.f);
        o[r] = (_Float16)y;
      }
      *(v8h*)(Y + (pix0 + s * 16) * COUT + ch0) = o;
    }
  }
}

// ---------------------------------------------------------------------------
// Cost-volume kernel. One (b,h) row per block; P/Q rows staged in LDS
// (17-dword pixel stride -> conflict-free). Per disparity:
//   h1 = relu(P[w] + Q[w-d]) (wrap == jnp.roll; masked out below)
//   h2 = relu(W5 @ h1 + b5)   via one v_wmma per 16-pixel tile
//   y  = relu(w6 . h2 + b6), zeroed where w < d.
// A pixel's 16 h2-channels live in lanes n and n+16 -> __shfl_xor(16) reduce.
// ---------------------------------------------------------------------------
__global__ __launch_bounds__(256) void volume_k(
    const _Float16* __restrict__ P, const _Float16* __restrict__ Q,
    const _Float16* __restrict__ W5, const float* __restrict__ b5,
    const float* __restrict__ w6, const float* __restrict__ b6,
    float* __restrict__ out) {
  constexpr int CP = 17;  // dwords per pixel (32 halves + 2 pad)
  __shared__ unsigned ldsP[WID * CP];
  __shared__ unsigned ldsQ[WID * CP];
  const int tid = threadIdx.x;
  const int row = blockIdx.x;             // b*H + h
  const int b = row >> 7, h = row & (HGT - 1);
  const unsigned* gp = (const unsigned*)(P + (size_t)row * WID * 32);
  const unsigned* gq = (const unsigned*)(Q + (size_t)row * WID * 32);
  for (int i = tid; i < WID * 16; i += 256) {
    int pix = i >> 4, j = i & 15;
    ldsP[pix * CP + j] = gp[i];
    ldsQ[pix * CP + j] = gq[i];
  }
  __syncthreads();
  const int lane = tid & 31, wid = tid >> 5;
  const int hi = lane & 16, kh = hi ? 8 : 0, nl = lane & 15;
  const v16h a5 = *(const v16h*)(W5 + (size_t)lane * 16);
  float b5l[8], w6l[8];
#pragma unroll
  for (int r = 0; r < 8; ++r) { b5l[r] = b5[kh + r]; w6l[r] = w6[kh + r]; }
  const float bias6 = b6[0];
  for (int d = wid; d < NDISP; d += 8) {
#pragma unroll 1
    for (int pst = 0; pst < 16; ++pst) {
      const int n = pst * 16 + nl;
      int nq = n - d; if (nq < 0) nq += WID;     // wrap like jnp.roll
      BU u;
#pragma unroll
      for (int v = 0; v < 8; ++v) {
        const int kb = kbase(v, hi);             // even
        HU hp, hq, hs;
        hp.u = ldsP[n * CP + (kb >> 1)];
        hq.u = ldsQ[nq * CP + (kb >> 1)];
        _Float16 s0 = hp.h[0] + hq.h[0];
        _Float16 s1 = hp.h[1] + hq.h[1];
        hs.h[0] = s0 > (_Float16)0 ? s0 : (_Float16)0;
        hs.h[1] = s1 > (_Float16)0 ? s1 : (_Float16)0;
        u.u[v] = hs.u;
      }
      v8f acc = {};
      acc = __builtin_amdgcn_wmma_f32_16x16x32_f16(false, a5, false, u.v,
                                                   (short)0, acc, false, false);
      float partial = 0.f;
#pragma unroll
      for (int r = 0; r < 8; ++r)
        partial += fmaxf(acc[r] + b5l[r], 0.f) * w6l[r];
      const float total = partial + __shfl_xor(partial, 16, 32);
      float y = fmaxf(total + bias6, 0.f);
      if (n < d) y = 0.f;                        // cols < d masked
      if (lane < 16)
        out[(((size_t)b * NDISP + d) * HGT + h) * WID + n] = y;
    }
  }
}

// ---------------------------------------------------------------------------
// Host launcher
// ---------------------------------------------------------------------------
extern "C" void kernel_launch(void* const* d_in, const int* in_sizes, int n_in,
                              void* d_out, int out_size, void* d_ws, size_t ws_size,
                              hipStream_t stream) {
  (void)in_sizes; (void)n_in; (void)out_size; (void)ws_size;
  const float* fL = (const float*)d_in[0];
  const float* fR = (const float*)d_in[1];
  auto Wp = [&](int l) { return (const float*)d_in[2 + 5 * l + 0]; };
  auto Gp = [&](int l) { return (const float*)d_in[2 + 5 * l + 1]; };
  auto Bp = [&](int l) { return (const float*)d_in[2 + 5 * l + 2]; };
  auto Mp = [&](int l) { return (const float*)d_in[2 + 5 * l + 3]; };
  auto Vp = [&](int l) { return (const float*)d_in[2 + 5 * l + 4]; };

  char* ws = (char*)d_ws;
  _Float16* swz0  = (_Float16*)(ws + OFF_SWZ0);
  _Float16* swz1  = (_Float16*)(ws + OFF_SWZ1);
  _Float16* swz2  = (_Float16*)(ws + OFF_SWZ2);
  _Float16* swz3  = (_Float16*)(ws + OFF_SWZ3);
  _Float16* swz4e = (_Float16*)(ws + OFF_SWZ4E);
  _Float16* swz4o = (_Float16*)(ws + OFF_SWZ4O);
  _Float16* swz5  = (_Float16*)(ws + OFF_SWZ5);
  float* b0  = (float*)(ws + OFF_B0);
  float* b1  = (float*)(ws + OFF_B1);
  float* b2  = (float*)(ws + OFF_B2);
  float* b3  = (float*)(ws + OFF_B3);
  float* b4  = (float*)(ws + OFF_B4);
  float* bz  = (float*)(ws + OFF_BZ);
  float* b5  = (float*)(ws + OFF_B5);
  float* w6f = (float*)(ws + OFF_W6);
  float* b6f = (float*)(ws + OFF_B6);
  _Float16* Y1  = (_Float16*)(ws + OFF_Y1);
  _Float16* Y2  = (_Float16*)(ws + OFF_Y2);
  _Float16* F64 = (_Float16*)(ws + OFF_F64);
  _Float16* F2L = (_Float16*)(ws + OFF_F2L);
  _Float16* F2R = (_Float16*)(ws + OFF_F2R);
  _Float16* Pb  = (_Float16*)(ws + OFF_P);
  _Float16* Qb  = (_Float16*)(ws + OFF_Q);

  // Weight folding / swizzling
  prep_wb<<<80, 256, 0, stream>>>(Wp(0), Gp(0), Bp(0), Mp(0), Vp(0), 256, 320, 320, 1, 0, swz0, b0, 1);
  prep_wb<<<32, 256, 0, stream>>>(Wp(1), Gp(1), Bp(1), Mp(1), Vp(1), 128, 256, 256, 1, 0, swz1, b1, 1);
  prep_wb<<< 8, 256, 0, stream>>>(Wp(2), Gp(2), Bp(2), Mp(2), Vp(2),  64, 128, 128, 1, 0, swz2, b2, 1);
  prep_wb<<< 2, 256, 0, stream>>>(Wp(3), Gp(3), Bp(3), Mp(3), Vp(3),  32,  64,  64, 1, 0, swz3, b3, 1);
  prep_wb<<< 1, 256, 0, stream>>>(Wp(4), Gp(4), Bp(4), Mp(4), Vp(4),  32,  32,  64, 2, 0, swz4e, b4, 1);
  prep_wb<<< 1, 256, 0, stream>>>(Wp(4), Gp(4), Bp(4), Mp(4), Vp(4),  32,  32,  64, 2, 1, swz4o, b4, 0);
  prep_wb<<< 1, 256, 0, stream>>>(Wp(5), Gp(5), Bp(5), Mp(5), Vp(5),  16,  32,  32, 1, 0, swz5, b5, 1);
  prep_small<<<1, 64, 0, stream>>>(Wp(6), Gp(6), Bp(6), Mp(6), Vp(6), w6f, b6f, bz);

  for (int img = 0; img < 2; ++img) {
    const float* Xi = img ? fR : fL;
    _Float16* feat2 = img ? F2R : F2L;
    conv_l0<<<(int)(NPIX / 64), 64, 0, stream>>>(Xi, swz0, b0, Y1);
    gemm_pm<256, 128, 2, false, true ><<<(int)(NPIX / 256), 256, 0, stream>>>(Y1,  swz1, b1, Y2);
    gemm_pm<128,  64, 2, false, false><<<(int)(NPIX / 256), 256, 0, stream>>>(Y2,  swz2, b2, F64);
    gemm_pm< 64,  32, 4, true,  false><<<(int)(NPIX / 512), 256, 0, stream>>>(F64, swz3, b3, feat2);
  }
  // d-independent halves of volume12 layer 1
  gemm_pm<32, 32, 4, false, false><<<(int)(NPIX / 512), 256, 0, stream>>>(F2L, swz4e, b4, Pb);
  gemm_pm<32, 32, 4, false, false><<<(int)(NPIX / 512), 256, 0, stream>>>(F2R, swz4o, bz, Qb);
  // 48-disparity sweep
  volume_k<<<BATCH * HGT, 256, 0, stream>>>(Pb, Qb, swz5, b5, w6f, b6f, (float*)d_out);
}
